// FlashSlidingWindowAttention_36412732735762
// MI455X (gfx1250) — matrix operand order repaired and checked by reading, not compiled, not run on hardware
//
#include <hip/hip_runtime.h>
#include <hip/hip_bf16.h>
#include <cstdint>
#include <cstddef>

// ---------------------------------------------------------------------------
// CDNA5 (gfx1250) sliding-window flash attention.
// bf16 WMMA compute, double-buffered TDM (tensor_load_to_lds) tile staging.
// B=2, S=2048, D=1024, H=16, d=64, window=512.
// ---------------------------------------------------------------------------

typedef __attribute__((ext_vector_type(16))) __bf16 v16bf;
typedef __attribute__((ext_vector_type(8)))  __bf16 v8bf;
typedef __attribute__((ext_vector_type(8)))  float  v8f;
typedef __attribute__((ext_vector_type(4)))  unsigned int v4u;
typedef __attribute__((ext_vector_type(8)))  int v8i;
typedef __attribute__((ext_vector_type(4)))  int v4i;

__device__ __forceinline__ v16bf cat8(v8bf lo, v8bf hi) {
  return __builtin_shufflevector(lo, hi, 0,1,2,3,4,5,6,7,8,9,10,11,12,13,14,15);
}

__device__ __forceinline__ v8f wmma_bf16(v16bf a, v16bf b, v8f c) {
  // D = A(16x32 bf16) * B(32x16 bf16) + C(16x16 f32)
  return __builtin_amdgcn_wmma_f32_16x16x32_bf16(
      /*neg_a=*/false, a, /*neg_b=*/false, b,
      /*c_mod=*/(short)0, c, /*reuse_a=*/false, /*reuse_b=*/false);
}

__device__ __forceinline__ float redmax16(float v) {
  v = fmaxf(v, __shfl_xor(v, 1, 16));
  v = fmaxf(v, __shfl_xor(v, 2, 16));
  v = fmaxf(v, __shfl_xor(v, 4, 16));
  v = fmaxf(v, __shfl_xor(v, 8, 16));
  return v;
}
__device__ __forceinline__ float redsum16(float v) {
  v += __shfl_xor(v, 1, 16);
  v += __shfl_xor(v, 2, 16);
  v += __shfl_xor(v, 4, 16);
  v += __shfl_xor(v, 8, 16);
  return v;
}

// ---------------------------------------------------------------------------
// Tensor Data Mover: 2D bf16 tile (tile_w x tile_h) from global (row stride
// `stride` elems) into LDS at byte offset lds_addr (rows stored contiguously).
// D# layout per CDNA5 ISA. Tracked on TENSORcnt.
// ---------------------------------------------------------------------------
__device__ __forceinline__ void tdm_load_2d_bf16(unsigned lds_addr,
                                                 const __bf16* gsrc,
                                                 unsigned tile_w, unsigned tile_h,
                                                 unsigned stride) {
  unsigned long long ga = (unsigned long long)(uintptr_t)gsrc;
  v4u g0;
  g0[0] = 1u;                                   // count=1, user descriptor
  g0[1] = lds_addr;                             // LDS byte address
  g0[2] = (unsigned)ga;                         // global_addr[31:0]
  g0[3] = ((unsigned)(ga >> 32) & 0x01FFFFFFu)  // global_addr[56:32]
          | 0x80000000u;                        // type=2 ("image")
  v8i g1;
  g1[0] = 0x00010000;                           // wg_mask=0, data_size=1 (2B)
  g1[1] = (int)((tile_w & 0xFFFFu) << 16);      // tensor_dim0[15:0] @ [63:48]
  g1[2] = (int)(((tile_w >> 16) & 0xFFFFu) |    // tensor_dim0[31:16]
                ((tile_h & 0xFFFFu) << 16));    // tensor_dim1[15:0] @ [95:80]
  g1[3] = (int)(((tile_h >> 16) & 0xFFFFu) |    // tensor_dim1[31:16]
                ((tile_w & 0xFFFFu) << 16));    // tile_dim0 @ [127:112]
  g1[4] = (int)(tile_h & 0xFFFFu);              // tile_dim1 @ [143:128]
  g1[5] = (int)stride;                          // tensor_dim0_stride[31:0]
  g1[6] = 0;                                    // stride[47:32], dim1_stride lo
  g1[7] = 0;
  v4i gz = {0, 0, 0, 0};
#if defined(__clang_major__) && (__clang_major__ >= 23)
  v8i gz8 = {0, 0, 0, 0, 0, 0, 0, 0};
  __builtin_amdgcn_tensor_load_to_lds(g0, g1, gz, gz, gz8, 0);
#else
  __builtin_amdgcn_tensor_load_to_lds(g0, g1, gz, gz, 0);
#endif
}

// ---------------------------------------------------------------------------
// fp32 -> bf16 conversion
// ---------------------------------------------------------------------------
__global__ void f32_to_bf16_kernel(const float* __restrict__ in,
                                   __bf16* __restrict__ out, int n) {
  int i = blockIdx.x * blockDim.x + threadIdx.x;
  if (i < n) out[i] = (__bf16)in[i];
}

// ---------------------------------------------------------------------------
// C[m][n] = sum_k A[m][k] * W[n][k]   (A: MxK bf16 row-major, W: NxK bf16)
// Block tile 128x64, 8 waves (wave32), each wave 16 rows x 64 cols.
// Double-buffered TDM staging: wave 0 issues tile k+1 while all waves run
// WMMA on tile k; TENSORcnt drained just before the single barrier.
// MODE 0: store bf16 into (B,H,S,d) head layout.  MODE 1: store f32 flat.
// ---------------------------------------------------------------------------
template <int MODE>
__global__ __launch_bounds__(256)
void gemm_xWT_kernel(const __bf16* __restrict__ A, const __bf16* __restrict__ W,
                     void* __restrict__ dst, int M, int N, int K, int S, int H) {
  __shared__ __align__(32) __bf16 As[2][128][32];  // 16 KB
  __shared__ __align__(32) __bf16 Ws[2][64][32];   //  8 KB

  const int m0 = blockIdx.x * 128;
  const int n0 = blockIdx.y * 64;
  const int tid  = threadIdx.x;
  const int wave = tid >> 5;
  const int lane = tid & 31;
  const int hf   = lane >> 4;   // which half of the wave (lane group)
  const int r    = lane & 15;

  const unsigned asBase[2] = {(unsigned)(uintptr_t)&As[0][0][0],
                              (unsigned)(uintptr_t)&As[1][0][0]};
  const unsigned wsBase[2] = {(unsigned)(uintptr_t)&Ws[0][0][0],
                              (unsigned)(uintptr_t)&Ws[1][0][0]};

  v8f acc[4] = {};

  // Prologue: stage k-tile 0 into buffer 0.
  if (wave == 0) {
    tdm_load_2d_bf16(asBase[0], &A[(size_t)m0 * K], 32, 128, (unsigned)K);
    tdm_load_2d_bf16(wsBase[0], &W[(size_t)n0 * K], 32, 64,  (unsigned)K);
    __builtin_amdgcn_s_wait_tensorcnt(0);
  }
  __syncthreads();

  for (int k0 = 0; k0 < K; k0 += 32) {
    const int  cur  = (k0 >> 5) & 1;
    const bool more = (k0 + 32 < K);
    if (wave == 0 && more) {  // stage next tile while computing this one
      tdm_load_2d_bf16(asBase[cur ^ 1], &A[(size_t)m0 * K + k0 + 32], 32, 128, (unsigned)K);
      tdm_load_2d_bf16(wsBase[cur ^ 1], &W[(size_t)n0 * K + k0 + 32], 32, 64,  (unsigned)K);
    }

    // A fragment (16x32 bf16): lane r = row, element i -> k = hf*16 + i.
    const __bf16* ap = &As[cur][wave * 16 + r][0];
    v16bf afrag = *(const v16bf*)(ap + hf * 16);
#pragma unroll
    for (int c = 0; c < 4; ++c) {
      // B fragment (32x16): element i -> B[k = hf*16 + i][n = r] = W[n][k]
      v16bf bfrag = *(const v16bf*)&Ws[cur][c * 16 + r][hf * 16];
      acc[c] = wmma_bf16(afrag, bfrag, acc[c]);
    }

    if (wave == 0 && more) __builtin_amdgcn_s_wait_tensorcnt(0);
    __syncthreads();
  }

  // Epilogue: C fragment element j -> row m0+wave*16 + j + 8*hf, col n0+c*16+r
#pragma unroll
  for (int c = 0; c < 4; ++c) {
    const int n = n0 + c * 16 + r;
#pragma unroll
    for (int j = 0; j < 8; ++j) {
      const int m = m0 + wave * 16 + j + 8 * hf;
      const float v = acc[c][j];
      if (MODE == 0) {
        const int b = m / S, s = m % S, h = n >> 6, dc = n & 63;
        ((__bf16*)dst)[(((size_t)b * H + h) * S + s) * 64 + dc] = (__bf16)v;
      } else {
        ((float*)dst)[(size_t)m * N + n] = v;
      }
    }
  }
}

// ---------------------------------------------------------------------------
// Flash sliding-window attention. Q,K,V in (B*H, S, 64) bf16.
// Block: 128 query rows (8 waves x 16). Key tiles of 32, double-buffered:
// K staged by TDM, V staged transposed by VALU, both one tile ahead.
// O written (B, S, H*64) bf16 for the final GEMM.
// ---------------------------------------------------------------------------
__global__ __launch_bounds__(256)
void attn_swa_kernel(const __bf16* __restrict__ Q, const __bf16* __restrict__ Kt,
                     const __bf16* __restrict__ V, __bf16* __restrict__ O) {
  constexpr int S = 2048, WIN = 512, H = 16;
  __shared__ __align__(32) __bf16 Ks[2][32][64];    // K tiles (TDM)      8 KB
  __shared__ __align__(32) __bf16 Vt[2][64][48];    // V tiles [d][t]    12 KB
  __shared__ __align__(32) __bf16 Ps[8][16][48];    // per-wave P        12 KB

  const int r0 = blockIdx.x * 128;
  const int h  = blockIdx.y;
  const int b  = blockIdx.z;
  const size_t bh = ((size_t)b * H + h) * S;

  const int tid  = threadIdx.x;
  const int wave = tid >> 5;
  const int lane = tid & 31;
  const int hf   = lane >> 4;
  const int r    = lane & 15;
  const int qr0  = r0 + wave * 16;

  const unsigned ksBase[2] = {(unsigned)(uintptr_t)&Ks[0][0][0],
                              (unsigned)(uintptr_t)&Ks[1][0][0]};

  // Q fragments for k0=0 and k0=32 (reused across all key tiles)
  const __bf16* qrow = Q + (bh + (size_t)(qr0 + r)) * 64;
  v16bf qa0 = *(const v16bf*)(qrow + hf * 16);
  v16bf qa1 = *(const v16bf*)(qrow + 32 + hf * 16);

  float mrow[8], lrow[8];
  v8f acc[4] = {};
#pragma unroll
  for (int j = 0; j < 8; ++j) { mrow[j] = -3.0e38f; lrow[j] = 0.0f; }

  int c_start = r0 - (WIN - 1);
  if (c_start < 0) c_start = 0;
  c_start &= ~31;
  const int c_end = r0 + 127;
  const float scale = 0.125f;  // d^-0.5, d=64

  const int srow = tid >> 3;          // 0..31: staging row
  const int scol = (tid & 7) * 8;     // 0..56: staging col chunk

  // Prologue: stage key tile c_start into buffer 0.
  {
    v8bf vv = *(const v8bf*)(V + (bh + (size_t)(c_start + srow)) * 64 + scol);
#pragma unroll
    for (int i = 0; i < 8; ++i) Vt[0][scol + i][srow] = vv[i];
    if (wave == 0) {
      tdm_load_2d_bf16(ksBase[0], Kt + (bh + (size_t)c_start) * 64, 64, 32, 64);
      __builtin_amdgcn_s_wait_tensorcnt(0);
    }
  }
  __syncthreads();

  for (int c0 = c_start; c0 <= c_end; c0 += 32) {
    const int  cur  = ((c0 - c_start) >> 5) & 1;
    const bool more = (c0 + 32 <= c_end);

    // Stage next key tile into the other buffer while computing this one.
    if (more) {
      v8bf vv = *(const v8bf*)(V + (bh + (size_t)(c0 + 32 + srow)) * 64 + scol);
#pragma unroll
      for (int i = 0; i < 8; ++i) Vt[cur ^ 1][scol + i][srow] = vv[i];
      if (wave == 0) {
        tdm_load_2d_bf16(ksBase[cur ^ 1], Kt + (bh + (size_t)(c0 + 32)) * 64, 64, 32, 64);
      }
      if (c0 + 64 <= c_end) {  // prefetch tile after next (global_prefetch_b8)
        __builtin_prefetch(Kt + (bh + (size_t)(c0 + 64 + srow)) * 64 + scol, 0, 1);
      }
    }

    const bool active = (c0 <= qr0 + 15) && (c0 + 31 >= qr0 - (WIN - 1));
    if (active) {
      // --- scores: two 16x16 subtiles over d=64 (2 chained WMMAs each) ---
      v8f s0 = {}, s1 = {};
      {
        v16bf b00 = *(const v16bf*)&Ks[cur][r][hf * 16];
        v16bf b01 = *(const v16bf*)&Ks[cur][r][32 + hf * 16];
        s0 = wmma_bf16(qa0, b00, s0);
        s0 = wmma_bf16(qa1, b01, s0);
        v16bf b10 = *(const v16bf*)&Ks[cur][16 + r][hf * 16];
        v16bf b11 = *(const v16bf*)&Ks[cur][16 + r][32 + hf * 16];
        s1 = wmma_bf16(qa0, b10, s1);
        s1 = wmma_bf16(qa1, b11, s1);
      }

      // --- online softmax over the 32-wide tile ---
      float alpha[8];
#pragma unroll
      for (int j = 0; j < 8; ++j) {
        const int row  = qr0 + j + 8 * hf;
        const int col0 = c0 + r;
        const int col1 = c0 + 16 + r;
        const bool ok0 = (col0 <= row) && (row - col0 < WIN);
        const bool ok1 = (col1 <= row) && (row - col1 < WIN);
        const float v0 = ok0 ? s0[j] * scale : -3.0e38f;
        const float v1 = ok1 ? s1[j] * scale : -3.0e38f;
        const float tmax = redmax16(fmaxf(v0, v1));
        const float mn = fmaxf(mrow[j], tmax);
        const float a  = __expf(mrow[j] - mn);
        const float p0 = ok0 ? __expf(v0 - mn) : 0.0f;
        const float p1 = ok1 ? __expf(v1 - mn) : 0.0f;
        lrow[j] = lrow[j] * a + redsum16(p0 + p1);
        mrow[j] = mn;
        alpha[j] = a;
        // stage P (C-fragment layout -> row-major LDS)
        Ps[wave][j + 8 * hf][r]      = (__bf16)p0;
        Ps[wave][j + 8 * hf][16 + r] = (__bf16)p1;
      }

      // rescale accumulator before adding this tile's P*V
#pragma unroll
      for (int c = 0; c < 4; ++c)
#pragma unroll
        for (int j = 0; j < 8; ++j) acc[c][j] *= alpha[j];

      // Same-wave LDS RAW: P stores above feed the A-fragment loads below.
      asm volatile("s_wait_dscnt 0x0" ::: "memory");

      // --- P (16x32) x V (32x64): 4 WMMAs ---
      const __bf16* pr = &Ps[wave][r][0];
      v16bf pa = *(const v16bf*)(pr + hf * 16);
#pragma unroll
      for (int c = 0; c < 4; ++c) {
        v16bf vb = *(const v16bf*)&Vt[cur][c * 16 + r][hf * 16];
        acc[c] = wmma_bf16(pa, vb, acc[c]);
      }
    }

    if (wave == 0 && more) __builtin_amdgcn_s_wait_tensorcnt(0);
    __syncthreads();
  }

  // Epilogue: O[b][s][h*64 + dc] = acc / l
#pragma unroll
  for (int c = 0; c < 4; ++c) {
#pragma unroll
    for (int j = 0; j < 8; ++j) {
      const int row = qr0 + j + 8 * hf;
      const float denom = lrow[j] > 1e-20f ? lrow[j] : 1e-20f;
      O[((size_t)b * S + row) * 1024 + h * 64 + c * 16 + r] =
          (__bf16)(acc[c][j] / denom);
    }
  }
}

// ---------------------------------------------------------------------------
// Launch
// ---------------------------------------------------------------------------
extern "C" void kernel_launch(void* const* d_in, const int* in_sizes, int n_in,
                              void* d_out, int out_size, void* d_ws, size_t ws_size,
                              hipStream_t stream) {
  (void)in_sizes; (void)n_in; (void)out_size; (void)ws_size;
  constexpr int B = 2, S = 2048, D = 1024, H = 16;
  constexpr size_t XE = (size_t)B * S * D;   // 4,194,304 elems (x/Q/K/V/O)
  constexpr size_t WE = (size_t)D * D;       // 1,048,576 elems per weight

  const float* x  = (const float*)d_in[0];
  const float* Wq = (const float*)d_in[1];
  const float* Wk = (const float*)d_in[2];
  const float* Wv = (const float*)d_in[3];
  const float* Wo = (const float*)d_in[4];

  __bf16* xb = (__bf16*)d_ws;
  __bf16* wq = xb + XE;
  __bf16* wk = wq + WE;
  __bf16* wv = wk + WE;
  __bf16* wo = wv + WE;
  __bf16* Qw = wo + WE;
  __bf16* Kw = Qw + XE;
  __bf16* Vw = Kw + XE;
  __bf16* Ow = Vw + XE;   // total ~50.3 MB of workspace

  // 1) fp32 -> bf16 conversions
  f32_to_bf16_kernel<<<(XE + 255) / 256, 256, 0, stream>>>(x, xb, (int)XE);
  f32_to_bf16_kernel<<<(WE + 255) / 256, 256, 0, stream>>>(Wq, wq, (int)WE);
  f32_to_bf16_kernel<<<(WE + 255) / 256, 256, 0, stream>>>(Wk, wk, (int)WE);
  f32_to_bf16_kernel<<<(WE + 255) / 256, 256, 0, stream>>>(Wv, wv, (int)WE);
  f32_to_bf16_kernel<<<(WE + 255) / 256, 256, 0, stream>>>(Wo, wo, (int)WE);

  // 2) Q/K/V projections: (B*S x D) * W^T -> (B,H,S,d) bf16
  dim3 ggrid(B * S / 128, D / 64, 1);
  gemm_xWT_kernel<0><<<ggrid, 256, 0, stream>>>(xb, wq, Qw, B * S, D, D, S, H);
  gemm_xWT_kernel<0><<<ggrid, 256, 0, stream>>>(xb, wk, Kw, B * S, D, D, S, H);
  gemm_xWT_kernel<0><<<ggrid, 256, 0, stream>>>(xb, wv, Vw, B * S, D, D, S, H);

  // 3) flash sliding-window attention -> O in (B, S, H*d) bf16
  attn_swa_kernel<<<dim3(S / 128, H, B), 256, 0, stream>>>(Qw, Kw, Vw, Ow);

  // 4) output projection: O * Wo^T -> fp32 d_out
  gemm_xWT_kernel<1><<<ggrid, 256, 0, stream>>>(Ow, wo, d_out, B * S, D, D, S, H);
}
